// STU_3393024163903
// MI455X (gfx1250) — matrix operand order repaired and checked by reading, not compiled, not run on hardware
//
#include <hip/hip_runtime.h>
#include <stdint.h>

// ---------------------------------------------------------------------------
// STU forward for MI455X (gfx1250, wave32, WMMA).
// Phases: pack -> [per-batch-chunk: spectral GEMM -> Toeplitz conv + AR] -> recurrence.
// All GEMMs on v_wmma_f32_16x16x32_bf16. Every B operand is kept in
// "fragment order" ([h][n][16 contiguous K-values]) so a B fragment is one
// aligned 32B load (2x b128) instead of 16 strided u16 gathers. Tile staging
// uses TDM tensor_load_to_lds (s_wait_tensorcnt), double-buffered in the conv.
// ---------------------------------------------------------------------------

#define B_  8
#define L_  2048
#define D_  256
#define K_  24
#define KU_ 3
#define KY_ 2
#define VP_ 64            // zero pad in front of each Vtilde column (causal masking)
#define VL_ (L_ + VP_)    // 2112
#define PANEL_ 8192       // one 32x256 B-stripe in fragment order: [2][256][16] bf16

typedef __bf16 bf16;
typedef __attribute__((ext_vector_type(16))) __bf16 v16bf;
typedef __attribute__((ext_vector_type(8)))  float  v8f;
typedef __attribute__((ext_vector_type(4)))  unsigned int u32x4;
typedef __attribute__((ext_vector_type(8)))  int i32x8;
typedef __attribute__((ext_vector_type(4)))  int i32x4;

#if defined(__has_builtin)
#if __has_builtin(__builtin_amdgcn_tensor_load_to_lds)
#define HAS_TDM 1
#endif
#endif
#ifndef HAS_TDM
#define HAS_TDM 0
#endif

#if HAS_TDM
#define TWAIT(n) __builtin_amdgcn_s_wait_tensorcnt(n)
#else
#define TWAIT(n)
#endif

// Flat shared-pointer low 32 bits == LDS byte address (LDS aperture lives in the
// high 32 bits of the flat address; ISA 10.2).
__device__ __forceinline__ unsigned lds_off(const void* p) {
  return (unsigned)(uintptr_t)p;
}

#if HAS_TDM
// 2D TDM load: tile of h rows x w elements (row stride = strideElems).
// D# group0: count=1, lds_addr, global_addr, type=2. group1: data_size, dims,
// strides. 6-arg builtin (clang-23 / therock-10.0 signature); groups 2-4 zero.
__device__ __forceinline__ void tdm_load_2d(unsigned lds, const void* g,
                                            unsigned w, unsigned h,
                                            unsigned strideElems, unsigned dszCode) {
  unsigned long long ga = (unsigned long long)(uintptr_t)g;
  u32x4 g0;
  g0[0] = 1u;                                              // count=1 (valid)
  g0[1] = lds;                                             // lds_addr [63:32]
  g0[2] = (unsigned)ga;                                    // global_addr[31:0]
  g0[3] = (unsigned)((ga >> 32) & 0x01FFFFFFu) | 0x80000000u; // addr[56:32] | type=2
  i32x8 g1;
  g1[0] = (int)(dszCode << 16);                            // data_size (0:1B 1:2B 2:4B)
  g1[1] = (int)((w & 0xFFFFu) << 16);                      // tensor_dim0 lo16
  g1[2] = (int)((w >> 16) | ((h & 0xFFFFu) << 16));        // tensor_dim0 hi | tensor_dim1 lo
  g1[3] = (int)((h >> 16) | ((w & 0xFFFFu) << 16));        // tensor_dim1 hi | tile_dim0
  g1[4] = (int)(h & 0xFFFFu);                              // tile_dim1, tile_dim2=0
  g1[5] = (int)strideElems;                                // tensor_dim0_stride lo32
  g1[6] = 0;
  g1[7] = 0;
  i32x4 z4 = {0, 0, 0, 0};
  i32x8 z8 = {0, 0, 0, 0, 0, 0, 0, 0};
  __builtin_amdgcn_tensor_load_to_lds(g0, g1, z4, z4, z8, 0);
}
#endif

// Stage a 2D tile into LDS. TDM path: wave 0 issues (wave-uniform branch via
// readfirstlane). Fallback: cooperative copy.
__device__ __forceinline__ void stage2d(bf16* dst, const bf16* src, int w, int h,
                                        int strideElems, int tid, int nthr) {
#if HAS_TDM
  if (__builtin_amdgcn_readfirstlane(tid >> 5) == 0)
    tdm_load_2d(lds_off(dst), src, (unsigned)w, (unsigned)h, (unsigned)strideElems, 1u);
  (void)nthr;
#else
  for (int i = tid; i < w * h; i += nthr)
    dst[i] = src[(size_t)(i / w) * strideElems + (i % w)];
#endif
}

// ----- WMMA fragment loaders (wave32; contiguous K order K = 16h + i, same for A and B) -----------
// A 16x32 bf16: lane m = lane&15; half h: VGPR j holds K = 2j+16h.
__device__ __forceinline__ v16bf load_a16(const bf16* t, int ld, int lane) {
  int m = lane & 15, h16 = (lane >> 4) * 16;
  v16bf a;
#pragma unroll
  for (int j = 0; j < 8; ++j) {
    int kk = 2 * j + h16;
    const bf16* p = t + (size_t)m * ld + kk;
    a[2 * j] = p[0];
    a[2 * j + 1] = p[1];
  }
  return a;
}
// Toeplitz A tile: A[m, kk] = vcol[base + m - kk] (reversed slice of a column).
__device__ __forceinline__ v16bf load_a_toep(const bf16* vcol, int base, int lane) {
  int m = lane & 15, h16 = (lane >> 4) * 16;
  v16bf a;
#pragma unroll
  for (int j = 0; j < 8; ++j) {
    int kk = 2 * j + h16;
    int idx = base + m - kk;
    a[2 * j] = vcol[idx];
    a[2 * j + 1] = vcol[idx - 1];
  }
  return a;
}
// B fragment from fragment-order panel [2][256][16]: lane (n,h) reads 32
// contiguous bytes -> 2x (ds|global)_load_b128. Element i holds K = 16h + i.
__device__ __forceinline__ v16bf load_bF(const bf16* panel, int n0, int lane) {
  int n = n0 + (lane & 15), h = lane >> 4;
  return *(const v16bf*)(panel + ((size_t)h * 256 + n) * 16);
}
__device__ __forceinline__ v8f wmma_bf16(v16bf a, v16bf b, v8f c) {
  return __builtin_amdgcn_wmma_f32_16x16x32_bf16(false, a, false, b, (short)0, c,
                                                 false, false);
}

// ----- pack kernels ---------------------------------------------------------
__global__ void k_cvt(const float* __restrict__ in, bf16* __restrict__ out, int n) {
  int i = blockIdx.x * 256 + threadIdx.x;
  if (i < n) out[i] = (bf16)in[i];
}
// Vt[k][VP+s] = eig_vals[k]^0.25 * eig_vecs[s,k];  Vt[k][0..VP) = 0 (causal pad)
__global__ void k_pack_V(const float* __restrict__ ev, const float* __restrict__ el,
                         bf16* __restrict__ Vt) {
  int i = blockIdx.x * 256 + threadIdx.x;
  if (i >= K_ * VL_) return;
  int k = i / VL_, j = i % VL_;
  float v = 0.0f;
  if (j >= VP_) v = ev[(size_t)(j - VP_) * K_ + k] * powf(el[k], 0.25f);
  Vt[i] = (bf16)v;
}
// Fragment-order repacks: out idx = (((blk8)*2 + h)*256 + o)*16 + i,
// K-value = blk*32 + 16h + i.
__global__ void k_pack_phiF(const float* __restrict__ mphi, bf16* __restrict__ out) {
  int idx = blockIdx.x * 256 + threadIdx.x;
  if (idx >= K_ * D_ * D_) return;
  int i = idx & 15, o = (idx >> 4) & 255, h = (idx >> 12) & 1;
  int dblk = (idx >> 13) & 7, k = idx >> 16;
  int d = dblk * 32 + h * 16 + i;
  out[idx] = (bf16)mphi[((size_t)k * D_ + d) * D_ + o];
}
__global__ void k_pack_muF(const float* __restrict__ mu, bf16* __restrict__ out) {
  int idx = blockIdx.x * 256 + threadIdx.x;
  if (idx >= KU_ * D_ * D_) return;
  int i = idx & 15, o = (idx >> 4) & 255, h = (idx >> 12) & 1;
  int dblk = (idx >> 13) & 7, iu = idx >> 16;
  int d = dblk * 32 + h * 16 + i;
  out[idx] = (bf16)mu[((size_t)o * D_ + d) * KU_ + iu];
}
__global__ void k_pack_myF(const float* __restrict__ my, bf16* __restrict__ out) {
  int idx = blockIdx.x * 256 + threadIdx.x;
  if (idx >= KY_ * D_ * D_) return;
  int i = idx & 15, o = (idx >> 4) & 255, h = (idx >> 12) & 1;
  int kb = (idx >> 13) & 7, mat = idx >> 16;
  int j = kb * 32 + h * 16 + i;
  out[idx] = (bf16)my[((size_t)o * KY_ + mat) * D_ + j];
}

// ----- phase 1: P[b][k][s][o] = sum_d u[b][s][d] * mphi[k][d][o] ------------
// grid (L/32, K, nb), 256 thr (8 waves x 32 output cols). Output written in
// panel-transposed fragment order: PF[b][k][panel = s0/32][h][o][i], s-local = 16h + i.
__global__ void __launch_bounds__(256)
k_gemm1(const bf16* __restrict__ u16b, const bf16* __restrict__ mphiF,
        bf16* __restrict__ PF) {
  extern __shared__ char smem[];
  bf16* ub  = (bf16*)smem;                         // [32][256]        16 KB
  bf16* mbF = (bf16*)(smem + 32 * 256 * 2);        // [8][2][256][16] 128 KB
  bf16* ob  = (bf16*)(smem + (32 * 256 + K_ * 0 + 256 * 256) * 2); // staging 16 KB
  const int tid = threadIdx.x, lane = tid & 31, wv = tid >> 5;
  const int s0 = blockIdx.x * 32, k = blockIdx.y, b = blockIdx.z;

  stage2d(ub, u16b + ((size_t)b * L_ + s0) * D_, 256, 32, 256, tid, 256);
  stage2d(mbF, mphiF + (size_t)k * D_ * D_, 8192, 8, 8192, tid, 256);
  TWAIT(0);
  __syncthreads();

  const int n0 = wv * 32;
  v8f acc[2][2] = {};
#pragma unroll
  for (int dblk = 0; dblk < 8; ++dblk) {
    v16bf a0 = load_a16(ub + dblk * 32, 256, lane);
    v16bf a1 = load_a16(ub + 16 * 256 + dblk * 32, 256, lane);
    const bf16* mg = mbF + (size_t)dblk * 2 * 256 * 16;
    v16bf b0 = load_bF(mg, n0, lane);
    v16bf b1 = load_bF(mg, n0 + 16, lane);
    acc[0][0] = wmma_bf16(a0, b0, acc[0][0]);
    acc[0][1] = wmma_bf16(a0, b1, acc[0][1]);
    acc[1][0] = wmma_bf16(a1, b0, acc[1][0]);
    acc[1][1] = wmma_bf16(a1, b1, acc[1][1]);
  }
  // D-fragment rows map straight into fragment order: element pos = r + 8*(lane>>4).
  const int nlo = lane & 15, hp = lane >> 4;
#pragma unroll
  for (int mt = 0; mt < 2; ++mt)
#pragma unroll
    for (int nt = 0; nt < 2; ++nt) {
      int col = n0 + nt * 16 + nlo;
#pragma unroll
      for (int r = 0; r < 8; ++r)
        ob[((size_t)mt * 256 + col) * 16 + r + 8 * hp] = (bf16)acc[mt][nt][r];
    }
  __syncthreads();
  // Coalesced 32B copies LDS -> global (fragment-order stripe, contiguous 16KB).
  v16bf* dst = (v16bf*)(PF + (((size_t)b * K_ + k) * (L_ / 32) + blockIdx.x) * PANEL_);
  const v16bf* srcv = (const v16bf*)ob;
  for (int g = tid; g < 512; g += 256) dst[g] = srcv[g];
}

// ----- phase 2: causal Toeplitz conv (spectral) + AR epilogue ---------------
// delta[b][t][o] = sum_{k, s<=t-2} Vt[k][t-2-s] PF[b][k][s][o]
//               + sum_{i<3, d}     u[b][t-i][d] BuF[i][d][o]
// grid (L/16, nb), 256 thr; K-dim flattened k-major so each 32-block is one k.
__global__ void __launch_bounds__(256)
k_conv(const bf16* __restrict__ PF, const bf16* __restrict__ Vt,
       const bf16* __restrict__ u16b, const bf16* __restrict__ BuF,
       float* __restrict__ delta) {
  extern __shared__ char smem[];
  bf16* vt  = (bf16*)smem;                               // [K][VL]    101376 B
  bf16* pbF = (bf16*)(smem + K_ * VL_ * 2);              // [2][PANEL]  32768 B
  bf16* uub = (bf16*)(smem + K_ * VL_ * 2 + 2 * PANEL_ * 2); // [18][256] 9216 B
  const int tid = threadIdx.x, lane = tid & 31, wv = tid >> 5;
  const int t0 = blockIdx.x * 16, b = blockIdx.y;
  const bf16* Pb = PF + (size_t)b * K_ * (L_ / 32) * PANEL_;

  stage2d(vt, Vt, VL_, K_, VL_, tid, 256);
  if (t0 >= 2) {
    stage2d(uub, u16b + ((size_t)b * L_ + (t0 - 2)) * D_, 256, 18, 256, tid, 256);
  } else {
    for (int i = tid; i < 2 * 256; i += 256) uub[i] = (bf16)0.0f;
    stage2d(uub + 2 * 256, u16b + (size_t)b * L_ * D_, 256, 16, 256, tid, 256);
  }
  const int nS = (t0 + 13) / 32 + 1;     // causal s-block count
  const int nQ = K_ * nS;
  stage2d(pbF, Pb, PANEL_, 1, PANEL_, tid, 256);   // (k=0, sb=0) stripe
  TWAIT(0);
  __syncthreads();

  const int n0 = wv * 32;
  v8f acc0 = {}, acc1 = {};
  for (int q = 0; q < nQ; ++q) {
    const int cur = q & 1;
    if (q + 1 < nQ) {   // double-buffered prefetch of next P stripe
      int k1 = (q + 1) / nS, sb1 = (q + 1) % nS;
      stage2d(pbF + (cur ^ 1) * PANEL_,
              Pb + ((size_t)k1 * (L_ / 32) + sb1) * PANEL_, PANEL_, 1, PANEL_,
              tid, 256);
      TWAIT(1);
    } else {
      TWAIT(0);
    }
    __syncthreads();
    const int k = q / nS, s0 = (q % nS) * 32;
    v16bf a  = load_a_toep(vt + (size_t)k * VL_, VP_ + t0 - 2 - s0, lane);
    v16bf b0 = load_bF(pbF + cur * PANEL_, n0, lane);
    v16bf b1 = load_bF(pbF + cur * PANEL_, n0 + 16, lane);
    acc0 = wmma_bf16(a, b0, acc0);
    acc1 = wmma_bf16(a, b1, acc1);
    __syncthreads();
  }

  // AR epilogue: 24 extra K-blocks; BuF fragments are coalesced 32B global loads.
#pragma unroll
  for (int i = 0; i < 3; ++i) {
#pragma unroll
    for (int dblk = 0; dblk < 8; ++dblk) {
      v16bf a = load_a16(uub + (2 - i) * 256 + dblk * 32, 256, lane);
      const bf16* bg = BuF + (size_t)(i * 8 + dblk) * 2 * 256 * 16;
      v16bf b0 = load_bF(bg, n0, lane);
      v16bf b1 = load_bF(bg, n0 + 16, lane);
      acc0 = wmma_bf16(a, b0, acc0);
      acc1 = wmma_bf16(a, b1, acc1);
    }
  }

  float* dout = delta + ((size_t)b * L_ + t0) * D_;
  const int nlo = lane & 15, mh = (lane >> 4) * 8;
#pragma unroll
  for (int r = 0; r < 8; ++r) {
    int m = r + mh;
    dout[(size_t)m * D_ + n0 + nlo] = acc0[r];
    dout[(size_t)m * D_ + n0 + 16 + nlo] = acc1[r];
  }
}

// ----- phase 3: recurrence y_t = M1 y_{t-1} + M2 y_{t-2} + delta_t ----------
// All 8 batch rows packed into one 16-row WMMA A operand; M1^T/M2^T fragment-
// order in LDS so every B fragment is one 32B ds load pair.
__global__ void __launch_bounds__(256)
k_recur(const bf16* __restrict__ MytF, const float* __restrict__ delta,
        float* __restrict__ yout) {
  extern __shared__ char smem[];
  bf16* mtF = (bf16*)smem;                     // [2][8][2][256][16] 256 KB
  bf16* yb  = (bf16*)(smem + 2 * D_ * D_ * 2); // [2][16][256]        16 KB
  const int tid = threadIdx.x, lane = tid & 31, wv = tid >> 5;
  stage2d(mtF, MytF, 8192, 16, 8192, tid, 256);
  for (int i = tid; i < 2 * 16 * 256; i += 256) yb[i] = (bf16)0.0f;
  TWAIT(0);
  __syncthreads();

  const int n0 = wv * 32, nlo = lane & 15, mh = (lane >> 4) * 8;
  for (int t = 0; t < L_; ++t) {
    const int p1 = (t + 1) & 1;  // y_{t-1}
    const int p2 = t & 1;        // y_{t-2}, becomes y_t
    v8f acc0 = {}, acc1 = {};
    if (mh == 0) {               // rows 0..7 = batches; rows 8..15 stay zero
#pragma unroll
      for (int r = 0; r < 8; ++r) {
        acc0[r] = delta[((size_t)r * L_ + t) * D_ + n0 + nlo];
        acc1[r] = delta[((size_t)r * L_ + t) * D_ + n0 + 16 + nlo];
      }
    }
    if (t + 1 < L_)
      __builtin_prefetch(delta + ((size_t)(lane & 7) * L_ + (t + 1)) * D_ + n0 + nlo, 0, 0);
#pragma unroll
    for (int kb = 0; kb < 8; ++kb) {
      v16bf a1 = load_a16(yb + p1 * 16 * 256 + kb * 32, 256, lane);
      v16bf a2 = load_a16(yb + p2 * 16 * 256 + kb * 32, 256, lane);
      const bf16* g1 = mtF + (size_t)kb * PANEL_;        // M1^T panel
      const bf16* g2 = mtF + (size_t)(8 + kb) * PANEL_;  // M2^T panel
      v16bf b10 = load_bF(g1, n0, lane);
      v16bf b11 = load_bF(g1, n0 + 16, lane);
      v16bf b20 = load_bF(g2, n0, lane);
      v16bf b21 = load_bF(g2, n0 + 16, lane);
      acc0 = wmma_bf16(a1, b10, acc0);
      acc1 = wmma_bf16(a1, b11, acc1);
      acc0 = wmma_bf16(a2, b20, acc0);
      acc1 = wmma_bf16(a2, b21, acc1);
    }
    __syncthreads();             // finish reads of yb[p2] before overwrite
    if (mh == 0) {
#pragma unroll
      for (int r = 0; r < 8; ++r) {
        yb[p2 * 16 * 256 + r * 256 + n0 + nlo] = (bf16)acc0[r];
        yb[p2 * 16 * 256 + r * 256 + n0 + 16 + nlo] = (bf16)acc1[r];
        yout[((size_t)r * L_ + t) * D_ + n0 + nlo] = acc0[r];
        yout[((size_t)r * L_ + t) * D_ + n0 + 16 + nlo] = acc1[r];
      }
    }
    __syncthreads();
  }
}

// ---------------------------------------------------------------------------
extern "C" void kernel_launch(void* const* d_in, const int* in_sizes, int n_in,
                              void* d_out, int out_size, void* d_ws, size_t ws_size,
                              hipStream_t stream) {
  const float* u     = (const float*)d_in[0];
  const float* m_y   = (const float*)d_in[1];
  const float* m_u   = (const float*)d_in[2];
  const float* m_phi = (const float*)d_in[3];
  const float* el    = (const float*)d_in[4];
  const float* ev    = (const float*)d_in[5];
  float* out = (float*)d_out;
  (void)in_sizes; (void)n_in; (void)out_size;

  char* ws = (char*)d_ws;
  size_t off = 0;
  auto carve = [&](size_t bytes) {
    char* p = ws + off;
    off = (off + bytes + 255) & ~(size_t)255;
    return p;
  };
  bf16*  u16   = (bf16*)carve((size_t)B_ * L_ * D_ * 2);   //  8.4 MB
  bf16*  mphiF = (bf16*)carve((size_t)K_ * D_ * D_ * 2);   //  3.1 MB
  bf16*  Vt    = (bf16*)carve((size_t)K_ * VL_ * 2);       //  0.1 MB
  bf16*  BuF   = (bf16*)carve((size_t)KU_ * D_ * D_ * 2);  //  0.4 MB
  bf16*  MytF  = (bf16*)carve((size_t)KY_ * D_ * D_ * 2);  //  0.3 MB
  float* delta = (float*)carve((size_t)B_ * L_ * D_ * 4);  // 16.8 MB

  // P is chunked by batch (25.2 MB/batch); chunk size adapts to ws_size.
  const size_t pPer = (size_t)K_ * L_ * D_ * 2;
  size_t remain = (ws_size > off) ? (ws_size - off) : 0;
  int nb = (int)(remain / pPer);
  if (nb < 1) nb = 1;
  if (nb > B_) nb = B_;
  bf16* PF = (bf16*)carve(pPer * (size_t)nb);

  // --- pack / convert ---
  {
    int nU = B_ * L_ * D_;
    k_cvt<<<(nU + 255) / 256, 256, 0, stream>>>(u, u16, nU);
    k_pack_phiF<<<(K_ * D_ * D_ + 255) / 256, 256, 0, stream>>>(m_phi, mphiF);
    k_pack_V<<<(K_ * VL_ + 255) / 256, 256, 0, stream>>>(ev, el, Vt);
    k_pack_muF<<<(KU_ * D_ * D_ + 255) / 256, 256, 0, stream>>>(m_u, BuF);
    k_pack_myF<<<(KY_ * D_ * D_ + 255) / 256, 256, 0, stream>>>(m_y, MytF);
  }

  const size_t ldsG1 = (32 * 256 + 256 * 256 + 32 * 256) * 2;            // 163840
  const size_t ldsCv = (K_ * VL_ + 2 * PANEL_ + 18 * 256) * 2;           // 143360
  const size_t ldsRe = (2 * D_ * D_ + 2 * 16 * 256) * 2;                 // 278528

  for (int b0 = 0; b0 < B_; b0 += nb) {
    int bs = (b0 + nb <= B_) ? nb : (B_ - b0);
    const bf16* ub = u16 + (size_t)b0 * L_ * D_;
    k_gemm1<<<dim3(L_ / 32, K_, bs), 256, ldsG1, stream>>>(ub, mphiF, PF);
    k_conv<<<dim3(L_ / 16, bs), 256, ldsCv, stream>>>(
        PF, Vt, ub, BuF, delta + (size_t)b0 * L_ * D_);
  }
  k_recur<<<1, 256, ldsRe, stream>>>(MytF, delta, out);
}
